// Transformer_89558658056431
// MI455X (gfx1250) — matrix operand order repaired and checked by reading, not compiled, not run on hardware
//
#include <hip/hip_runtime.h>
#include <hip/hip_bf16.h>
#include <math.h>

// ---------------------------------------------------------------------------
// Types for CDNA5 WMMA (wave32): V_WMMA_F32_16X16X32_BF16
// ---------------------------------------------------------------------------
typedef __attribute__((ext_vector_type(16))) __bf16 v16bf;
typedef __attribute__((ext_vector_type(8)))  float  v8f;

union FragBF {
    v16bf v;
    uint4 q[2];
};

// fp32x2 -> packed bf16x2 (RNE) in one VALU op. src0 -> low half.
__device__ __forceinline__ unsigned pack2(float a, float b) {
    unsigned r;
    asm("v_cvt_pk_bf16_f32 %0, %1, %2" : "=v"(r) : "v"(a), "v"(b));
    return r;
}

struct TrueT  { static constexpr bool value = true;  };
struct FalseT { static constexpr bool value = false; };

// ---------------------------------------------------------------------------
// Generic batched GEMM: C = act( alpha * A*B + bias ), with optional masking.
//   A: [M,K] row-major (lda), fp32
//   B: flags&1 ? [N,K] row-major (ldb)   (pre-transposed view: K^T, emb^T)
//              : [K,N] row-major (ldb)
//   C: [M,N] row-major (ldc), fp32
// Batched via blockIdx.z: ptr += (z/batchH)*Outer + (z%batchH)*Inner,
//                         bias += z*biasS
// maskMode: 0 none, 1 causal (col>row -> -1e30), 2 diag (col==row -> -1e30)
// flags: bit0 = B transposed, bit1 = ReLU
//
// 128xBNT block tile (BNT = 64 or 128), BK=32, 8 waves; per wave 2x(BNT/32)
// WMMA 16x16x32 bf16 tiles. Double-buffered LDS staging (fp32 -> bf16 via
// v_cvt_pk_bf16_f32) with a separate load phase so all global_load_b128
// issue back-to-back; global_prefetch lookahead; full-tile fast path.
// ---------------------------------------------------------------------------
#define BM 128
#define BK 32
#define LDS_STRIDE 40   // BK + 8 shorts of pad (80B rows, 16B aligned)

template <int BNT>
__global__ __launch_bounds__(256)
void wmma_gemm_bf16(const float* __restrict__ A, const float* __restrict__ Bm,
                    const float* __restrict__ bias, float* __restrict__ C,
                    int M, int N, int K, int lda, int ldb, int ldc, int batchH,
                    long long aO, long long aI, long long bO, long long bI,
                    long long cO, long long cI, long long biasS,
                    int flags, float alpha, int maskMode)
{
    constexpr int NI = BNT / 32;          // 16-col WMMA tiles per wave
    __shared__ unsigned short sA[2][BM  * LDS_STRIDE];
    __shared__ unsigned short sB[2][BNT * LDS_STRIDE];

    const int z = blockIdx.z;
    A  += (long long)(z / batchH) * aO + (long long)(z % batchH) * aI;
    Bm += (long long)(z / batchH) * bO + (long long)(z % batchH) * bI;
    C  += (long long)(z / batchH) * cO + (long long)(z % batchH) * cI;
    if (bias) bias += (long long)z * biasS;

    const int m0   = blockIdx.y * BM;
    const int n0   = blockIdx.x * BNT;
    const int tid  = threadIdx.x;
    const int lane = tid & 31;
    const int wave = tid >> 5;
    const int wm   = wave & 3;   // 4 waves along M (32 rows each)
    const int wn   = wave >> 2;  // 2 waves along N (BNT/2 cols each)

    // Stage one BMxBK A-tile and a B-tile into LDS, converting fp32 -> bf16.
    // Phase 1 issues every global load (and prefetch); phase 2 converts and
    // stores to LDS. B lands as [N][K] either way.
    auto stage = [&](int k0, unsigned short* dA, unsigned short* dB, auto FULL) {
        constexpr bool FB = decltype(FULL)::value;
        const bool pf = (k0 + BK) < K;
        float4 fa[4], fb[NI];

        #pragma unroll
        for (int it = 0; it < 4; ++it) {               // A loads: 1024 float4
            int idx = tid + it * 256;
            int row = idx >> 3;
            int c4  = (idx & 7) << 2;
            int gr  = m0 + row;
            fa[it] = make_float4(0.f, 0.f, 0.f, 0.f);
            if (FB || gr < M) {
                const float* p = A + (size_t)gr * lda + k0 + c4;
                fa[it] = *(const float4*)p;
                if (pf) __builtin_prefetch(p + BK, 0, 3);   // global_prefetch
            }
        }
        if (flags & 1) {
            #pragma unroll
            for (int it = 0; it < NI; ++it) {          // B loads ([N,K] direct)
                int idx = tid + it * 256;
                int row = idx >> 3;
                int c4  = (idx & 7) << 2;
                int gr  = n0 + row;
                fb[it] = make_float4(0.f, 0.f, 0.f, 0.f);
                if (FB || gr < N) {
                    const float* p = Bm + (size_t)gr * ldb + k0 + c4;
                    fb[it] = *(const float4*)p;
                    if (pf) __builtin_prefetch(p + BK, 0, 3);
                }
            }
        } else {
            constexpr int CPR = BNT / 4;               // float4 per K-row
            #pragma unroll
            for (int it = 0; it < NI; ++it) {          // B loads ([K,N] rows)
                int idx = tid + it * 256;
                int kk  = idx / CPR;
                int c4  = (idx % CPR) << 2;
                int gn  = n0 + c4;
                fb[it] = make_float4(0.f, 0.f, 0.f, 0.f);
                if (FB || gn < N) {
                    const float* p = Bm + (size_t)(k0 + kk) * ldb + gn;
                    fb[it] = *(const float4*)p;
                    if (pf) __builtin_prefetch(p + (size_t)BK * ldb, 0, 3);
                }
            }
        }

        // ---- phase 2: convert + LDS store ----
        #pragma unroll
        for (int it = 0; it < 4; ++it) {
            int idx = tid + it * 256;
            int row = idx >> 3;
            int c4  = (idx & 7) << 2;
            uint2 pk; pk.x = pack2(fa[it].x, fa[it].y); pk.y = pack2(fa[it].z, fa[it].w);
            *(uint2*)&dA[row * LDS_STRIDE + c4] = pk;
        }
        if (flags & 1) {
            #pragma unroll
            for (int it = 0; it < NI; ++it) {
                int idx = tid + it * 256;
                int row = idx >> 3;
                int c4  = (idx & 7) << 2;
                uint2 pk; pk.x = pack2(fb[it].x, fb[it].y); pk.y = pack2(fb[it].z, fb[it].w);
                *(uint2*)&dB[row * LDS_STRIDE + c4] = pk;
            }
        } else {
            constexpr int CPR = BNT / 4;
            #pragma unroll
            for (int it = 0; it < NI; ++it) {
                int idx = tid + it * 256;
                int kk  = idx / CPR;
                int c4  = (idx % CPR) << 2;
                unsigned pxy = pack2(fb[it].x, fb[it].y), pzw = pack2(fb[it].z, fb[it].w);
                dB[(c4 + 0) * LDS_STRIDE + kk] = (unsigned short)(pxy & 0xffffu);
                dB[(c4 + 1) * LDS_STRIDE + kk] = (unsigned short)(pxy >> 16);
                dB[(c4 + 2) * LDS_STRIDE + kk] = (unsigned short)(pzw & 0xffffu);
                dB[(c4 + 3) * LDS_STRIDE + kk] = (unsigned short)(pzw >> 16);
            }
        }
    };

    v8f acc[2][NI];
    {
        v8f zacc = {0.f,0.f,0.f,0.f,0.f,0.f,0.f,0.f};
        #pragma unroll
        for (int mi = 0; mi < 2; ++mi)
            #pragma unroll
            for (int ni = 0; ni < NI; ++ni) acc[mi][ni] = zacc;
    }

    const bool fullTile = (m0 + BM <= M) && (n0 + BNT <= N);
    const int  nt = (K + BK - 1) / BK;

    if (fullTile) stage(0, sA[0], sB[0], TrueT{});
    else          stage(0, sA[0], sB[0], FalseT{});
    __syncthreads();

    for (int i = 0; i < nt; ++i) {
        const int cur = i & 1;
        if (i + 1 < nt) {
            if (fullTile) stage((i + 1) * BK, sA[cur ^ 1], sB[cur ^ 1], TrueT{});
            else          stage((i + 1) * BK, sA[cur ^ 1], sB[cur ^ 1], FalseT{});
        }

        const unsigned short* cA = sA[cur];
        const unsigned short* cB = sB[cur];

        // B fragments: col N = lane&15, 16 consecutive K at (lane>>4)*16
        FragBF bfr[NI];
        #pragma unroll
        for (int ni = 0; ni < NI; ++ni) {
            int rb = (wn * (BNT/2) + ni * 16 + (lane & 15)) * LDS_STRIDE + ((lane >> 4) << 4);
            bfr[ni].q[0] = *(const uint4*)&cB[rb];
            bfr[ni].q[1] = *(const uint4*)&cB[rb + 8];
        }
        // A fragments: row M = lane&15, 16 consecutive K at (lane>>4)*16
        #pragma unroll
        for (int mi = 0; mi < 2; ++mi) {
            int ra = (wm * 32 + mi * 16 + (lane & 15)) * LDS_STRIDE + ((lane >> 4) << 4);
            FragBF af;
            af.q[0] = *(const uint4*)&cA[ra];
            af.q[1] = *(const uint4*)&cA[ra + 8];
            #pragma unroll
            for (int ni = 0; ni < NI; ++ni) {
                acc[mi][ni] = __builtin_amdgcn_wmma_f32_16x16x32_bf16(
                    false, af.v, false, bfr[ni].v, (short)0, acc[mi][ni], false, false);
            }
        }
        __syncthreads();
    }

    // ---- epilogue: alpha, bias, mask, relu ----
    #pragma unroll
    for (int mi = 0; mi < 2; ++mi) {
        #pragma unroll
        for (int ni = 0; ni < NI; ++ni) {
            int baseM = m0 + wm * 32 + mi * 16 + ((lane >> 4) << 3);
            int gN    = n0 + wn * (BNT/2) + ni * 16 + (lane & 15);
            if (gN >= N) continue;
            float bv = bias ? bias[gN] : 0.0f;
            #pragma unroll
            for (int r = 0; r < 8; ++r) {
                int gM = baseM + r;
                if (gM >= M) continue;
                float vv = acc[mi][ni][r] * alpha + bv;
                if (maskMode == 1) { if (gN > gM)  vv = -1e30f; }
                else if (maskMode == 2) { if (gN == gM) vv = -1e30f; }
                if (flags & 2) vv = fmaxf(vv, 0.0f);
                C[(size_t)gM * ldc + gN] = vv;
            }
        }
    }
}

// ---------------------------------------------------------------------------
// Row softmax, cols == 256 (8 per lane), one wave per row
// ---------------------------------------------------------------------------
__global__ __launch_bounds__(256)
void softmax256_kernel(float* __restrict__ s, int rows)
{
    int wid  = (int)((blockIdx.x * blockDim.x + threadIdx.x) >> 5);
    int lane = threadIdx.x & 31;
    if (wid >= rows) return;
    float* row = s + (size_t)wid * 256;
    float vals[8];
    float mx = -3.0e38f;
    #pragma unroll
    for (int i = 0; i < 8; ++i) { vals[i] = row[lane + i * 32]; mx = fmaxf(mx, vals[i]); }
    #pragma unroll
    for (int o = 16; o > 0; o >>= 1) mx = fmaxf(mx, __shfl_xor(mx, o, 32));
    float sum = 0.f;
    #pragma unroll
    for (int i = 0; i < 8; ++i) { vals[i] = __expf(vals[i] - mx); sum += vals[i]; }
    #pragma unroll
    for (int o = 16; o > 0; o >>= 1) sum += __shfl_xor(sum, o, 32);
    float inv = 1.0f / sum;
    #pragma unroll
    for (int i = 0; i < 8; ++i) row[lane + i * 32] = vals[i] * inv;
}

// ---------------------------------------------------------------------------
// Fused residual + LayerNorm, D == 512 (16 per lane), one wave per row
// out = LN(a + resid) * g + b ; out may alias resid (wave owns its row)
// ---------------------------------------------------------------------------
__global__ __launch_bounds__(256)
void ln_res_kernel(const float* __restrict__ a, const float* __restrict__ resid,
                   const float* __restrict__ g, const float* __restrict__ be,
                   float* __restrict__ out, int rows)
{
    int wid  = (int)((blockIdx.x * blockDim.x + threadIdx.x) >> 5);
    int lane = threadIdx.x & 31;
    if (wid >= rows) return;
    const float* pa = a     + (size_t)wid * 512;
    const float* pb = resid + (size_t)wid * 512;
    float vals[16];
    float sum = 0.f;
    #pragma unroll
    for (int i = 0; i < 16; ++i) {
        int c = lane + i * 32;
        float t = pa[c] + pb[c];
        vals[i] = t; sum += t;
    }
    #pragma unroll
    for (int o = 16; o > 0; o >>= 1) sum += __shfl_xor(sum, o, 32);
    float mean = sum * (1.0f / 512.0f);
    float vs = 0.f;
    #pragma unroll
    for (int i = 0; i < 16; ++i) { float d = vals[i] - mean; vs += d * d; }
    #pragma unroll
    for (int o = 16; o > 0; o >>= 1) vs += __shfl_xor(vs, o, 32);
    float inv = rsqrtf(vs * (1.0f / 512.0f) + 1e-5f);
    float* po = out + (size_t)wid * 512;
    #pragma unroll
    for (int i = 0; i < 16; ++i) {
        int c = lane + i * 32;
        po[c] = (vals[i] - mean) * inv * g[c] + be[c];
    }
}

// ---------------------------------------------------------------------------
// Token embedding * sqrt(D) + sinusoidal positional encoding.
// shift=1 implements the right-shifted decoder input (token 0 at s==0).
// One block per token row.
// ---------------------------------------------------------------------------
__global__ __launch_bounds__(128)
void embed_kernel(const int* __restrict__ tok, const float* __restrict__ emb,
                  float* __restrict__ out, int S, int D, float scale, int shift)
{
    int row = blockIdx.x;            // b*S + s
    int s   = row % S;
    int t;
    if (shift) t = (s == 0) ? 0 : tok[row - 1];
    else       t = tok[row];
    const float* e = emb + (size_t)t * D;
    const float neg_ln1e4_over_D = -9.2103403719761836f / (float)D;
    for (int c = threadIdx.x; c < D; c += blockDim.x) {
        int   i2  = c & ~1;
        float div = __expf((float)i2 * neg_ln1e4_over_D);
        float ang = (float)s * div;
        float pe  = (c & 1) ? __cosf(ang) : __sinf(ang);
        out[(size_t)row * D + c] = e[c] * scale + pe;
    }
}

// ---------------------------------------------------------------------------
// Host orchestration
// ---------------------------------------------------------------------------
namespace {
constexpr int Lh = 6, Dh = 512, HIDh = 2048, Vh = 32000, Hh = 8, Bb = 8, Sh = 256;
constexpr int MR = Bb * Sh;     // 2048 token rows (S == T)
constexpr int DK = Dh / Hh;     // 64
constexpr int FLAG_BT   = 1;
constexpr int FLAG_RELU = 2;

inline void gemm(hipStream_t st, const float* A, const float* B, const float* bias,
                 float* C, int M, int N, int K, int lda, int ldb, int ldc,
                 int batch, int batchH,
                 long long aO, long long aI, long long bO, long long bI,
                 long long cO, long long cI, long long biasS,
                 int flags, float alpha, int maskMode)
{
    const long long my = (M + BM - 1) / BM;
    const long long blocks128 = ((N + 127) / 128) * my * batch;
    // Fill the chip: fall back to the 64-wide tile when the launch is small.
    if (N <= 64 || blocks128 < 192) {
        dim3 grid((N + 63) / 64, (unsigned)my, batch);
        wmma_gemm_bf16<64><<<grid, dim3(256), 0, st>>>(A, B, bias, C, M, N, K,
            lda, ldb, ldc, batchH, aO, aI, bO, bI, cO, cI, biasS, flags, alpha, maskMode);
    } else {
        dim3 grid((N + 127) / 128, (unsigned)my, batch);
        wmma_gemm_bf16<128><<<grid, dim3(256), 0, st>>>(A, B, bias, C, M, N, K,
            lda, ldb, ldc, batchH, aO, aI, bO, bI, cO, cI, biasS, flags, alpha, maskMode);
    }
}
} // namespace

extern "C" void kernel_launch(void* const* d_in, const int* in_sizes, int n_in,
                              void* d_out, int out_size, void* d_ws, size_t ws_size,
                              hipStream_t stream)
{
    (void)in_sizes; (void)n_in; (void)out_size; (void)ws_size;

    const int*   src        = (const int*)  d_in[0];
    const int*   tgt        = (const int*)  d_in[1];
    const float* emb        = (const float*)d_in[2];
    const float* enc_attn_w = (const float*)d_in[3];
    const float* enc_attn_b = (const float*)d_in[4];
    const float* enc_w1     = (const float*)d_in[5];
    const float* enc_b1     = (const float*)d_in[6];
    const float* enc_w2     = (const float*)d_in[7];
    const float* enc_b2     = (const float*)d_in[8];
    const float* enc_ln_g   = (const float*)d_in[9];
    const float* enc_ln_b   = (const float*)d_in[10];
    const float* dec_sa_w   = (const float*)d_in[11];
    const float* dec_sa_b   = (const float*)d_in[12];
    const float* dec_ca_w   = (const float*)d_in[13];
    const float* dec_ca_b   = (const float*)d_in[14];
    const float* dec_w1     = (const float*)d_in[15];
    const float* dec_b1     = (const float*)d_in[16];
    const float* dec_w2     = (const float*)d_in[17];
    const float* dec_b2     = (const float*)d_in[18];
    const float* dec_ln_g   = (const float*)d_in[19];
    const float* dec_ln_b   = (const float*)d_in[20];

    float* ws = (float*)d_ws;
    const size_t MD = (size_t)MR * Dh;       // 1,048,576 floats
    float* x    = ws + 0 * MD;               // encoder activations [2048,512]
    float* y    = ws + 1 * MD;               // decoder activations
    float* q    = ws + 2 * MD;               // q,k,v MUST stay MD apart, in order
    float* k    = ws + 3 * MD;
    float* v    = ws + 4 * MD;
    float* attn = ws + 5 * MD;               // attention context [B,S,H,dk]
    float* tmp  = ws + 6 * MD;               // proj / ffn2 output
    float* big  = ws + 7 * MD;               // 4*MD: scores [64,256,256] OR ffn hidden [2048,2048]

    const float embScale = sqrtf((float)Dh);

    // ---- embeddings (+ positional encoding; decoder input right-shifted) ----
    embed_kernel<<<MR, 128, 0, stream>>>(src, emb, x, Sh, Dh, embScale, 0);
    embed_kernel<<<MR, 128, 0, stream>>>(tgt, emb, y, Sh, Dh, embScale, 1);

    auto run_ln = [&](const float* a, float* act, const float* g, const float* be) {
        ln_res_kernel<<<MR / 8, 256, 0, stream>>>(a, act, g, be, act, MR);
    };

    // Multi-head attention: projections (batched over weight index) ->
    // scores -> softmax -> P@V -> out proj
    auto run_attn = [&](const float* qin, const float* kvin,
                        const float* w, const float* bvec, float* outbuf, int maskMode) {
        if (qin == kvin) {
            // self-attention: one batch-3 launch covers Q,K,V projections
            gemm(stream, qin, w, bvec, q, MR, Dh, Dh, Dh, Dh, Dh,
                 3, 1, 0, 0, (long long)Dh * Dh, 0, (long long)MD, 0, Dh,
                 0, 1.0f, 0);
        } else {
            // cross-attention: Q from decoder state, batch-2 K,V from encoder
            gemm(stream, qin, w, bvec, q, MR, Dh, Dh, Dh, Dh, Dh,
                 1, 1, 0, 0, 0, 0, 0, 0, 0, 0, 1.0f, 0);
            gemm(stream, kvin, w + (size_t)Dh * Dh, bvec + Dh, k, MR, Dh, Dh, Dh, Dh, Dh,
                 2, 1, 0, 0, (long long)Dh * Dh, 0, (long long)MD, 0, Dh,
                 0, 1.0f, 0);
        }
        // scores[b,h] = (Q_h K_h^T) / sqrt(dk), masked.  batch = B*H
        gemm(stream, q, k, nullptr, big, Sh, Sh, DK, Dh, Dh, Sh,
             Bb * Hh, Hh,
             (long long)Sh * Dh, DK,                       // A strides (b, h)
             (long long)Sh * Dh, DK,                       // B strides (b, h)
             (long long)Hh * Sh * Sh, (long long)Sh * Sh,  // C strides
             0,
             FLAG_BT, 0.125f, maskMode);
        softmax256_kernel<<<(Bb * Hh * Sh) / 8, 256, 0, stream>>>(big, Bb * Hh * Sh);
        // context[b,:,h,:] = P @ V_h  (writes straight into [B,S,H*dk] layout)
        gemm(stream, big, v, nullptr, attn, Sh, DK, Sh, Sh, Dh, Dh,
             Bb * Hh, Hh,
             (long long)Hh * Sh * Sh, (long long)Sh * Sh,
             (long long)Sh * Dh, DK,
             (long long)Sh * Dh, DK,
             0,
             0, 1.0f, 0);
        gemm(stream, attn, w + 3 * (size_t)Dh * Dh, bvec + 3 * Dh, outbuf,
             MR, Dh, Dh, Dh, Dh, Dh,
             1, 1, 0, 0, 0, 0, 0, 0, 0, 0, 1.0f, 0);
    };

    auto run_ffn = [&](float* act, const float* w1, const float* b1,
                       const float* w2, const float* b2,
                       const float* g, const float* be) {
        gemm(stream, act, w1, b1, big, MR, HIDh, Dh, Dh, HIDh, HIDh,
             1, 1, 0, 0, 0, 0, 0, 0, 0, FLAG_RELU, 1.0f, 0);
        gemm(stream, big, w2, b2, tmp, MR, Dh, HIDh, HIDh, Dh, Dh,
             1, 1, 0, 0, 0, 0, 0, 0, 0, 0, 1.0f, 0);
        run_ln(tmp, act, g, be);
    };

    // ---- encoder stack (diag-masked self-attention per reference) ----
    for (int i = 0; i < Lh; ++i) {
        const float* w  = enc_attn_w + (size_t)i * 4 * Dh * Dh;
        const float* bb = enc_attn_b + (size_t)i * 4 * Dh;
        run_attn(x, x, w, bb, tmp, /*diag*/2);
        run_ln(tmp, x, enc_ln_g + (size_t)i * 2 * Dh, enc_ln_b + (size_t)i * 2 * Dh);
        run_ffn(x, enc_w1 + (size_t)i * Dh * HIDh, enc_b1 + (size_t)i * HIDh,
                enc_w2 + (size_t)i * HIDh * Dh,    enc_b2 + (size_t)i * Dh,
                enc_ln_g + (size_t)i * 2 * Dh + Dh, enc_ln_b + (size_t)i * 2 * Dh + Dh);
    }

    // ---- decoder stack ----
    for (int i = 0; i < Lh; ++i) {
        const float* wsa = dec_sa_w + (size_t)i * 4 * Dh * Dh;
        const float* bsa = dec_sa_b + (size_t)i * 4 * Dh;
        run_attn(y, y, wsa, bsa, tmp, /*causal*/1);
        run_ln(tmp, y, dec_ln_g + (size_t)i * 3 * Dh, dec_ln_b + (size_t)i * 3 * Dh);

        const float* wca = dec_ca_w + (size_t)i * 4 * Dh * Dh;
        const float* bca = dec_ca_b + (size_t)i * 4 * Dh;
        run_attn(y, x, wca, bca, tmp, /*none*/0);
        run_ln(tmp, y, dec_ln_g + (size_t)i * 3 * Dh + Dh, dec_ln_b + (size_t)i * 3 * Dh + Dh);

        run_ffn(y, dec_w1 + (size_t)i * Dh * HIDh, dec_b1 + (size_t)i * HIDh,
                dec_w2 + (size_t)i * HIDh * Dh,    dec_b2 + (size_t)i * Dh,
                dec_ln_g + (size_t)i * 3 * Dh + 2 * Dh, dec_ln_b + (size_t)i * 3 * Dh + 2 * Dh);
    }

    // ---- tied output projection: out = y @ emb^T  [2048 x 32000] ----
    gemm(stream, y, emb, nullptr, (float*)d_out, MR, Vh, Dh, Dh, Dh, Vh,
         1, 1, 0, 0, 0, 0, 0, 0, 0, FLAG_BT, 1.0f, 0);
}
